// mygat_5085241278874
// MI455X (gfx1250) — compile-verified, matrix operand order checked by reading, not run on hardware
//
#include <hip/hip_runtime.h>
#include <hip/hip_bf16.h>

#define N_NODES 50000
#define E_EDGES 800000
#define ET_EDGES (E_EDGES + N_NODES)   // with self loops
#define HEADS 4
#define NEG_SLOPE 0.2f

typedef __attribute__((ext_vector_type(16))) __bf16 v16bf;
typedef __attribute__((ext_vector_type(8)))  float  v8f;

union FragBF { v16bf v; uint4 u[2]; };

// ---------- helpers ----------
__device__ __forceinline__ unsigned short f2bf(float f) {
    unsigned int u = __float_as_uint(f);
    unsigned int r = u + 0x7FFFu + ((u >> 16) & 1u);   // RNE
    return (unsigned short)(r >> 16);
}

__device__ __forceinline__ void edge_nodes(const int* __restrict__ ei, int e, int& s, int& d) {
    if (e < E_EDGES) { s = ei[e]; d = ei[E_EDGES + e]; }
    else             { s = e - E_EDGES; d = s; }       // self loop
}

__device__ __forceinline__ float lrelu(float v) {
    return v > 0.0f ? v : NEG_SLOPE * v;
}

// ---------- kernels ----------
__global__ void k_cast_bf16(const float* __restrict__ src, unsigned short* __restrict__ dst, long n) {
    long i = (long)blockIdx.x * blockDim.x + threadIdx.x;
    if (i < n) dst[i] = f2bf(src[i]);
}

// W: [din, doutTotal] fp32 row-major -> Wt: [doutPad, din] bf16 (transposed, zero padded rows)
__global__ void k_pack_w(const float* __restrict__ W, unsigned short* __restrict__ Wt,
                         int din, int doutTotal, int doutPad) {
    int i = blockIdx.x * blockDim.x + threadIdx.x;
    int total = doutPad * din;
    if (i >= total) return;
    int n = i / din, k = i % din;
    float v = (n < doutTotal) ? W[(long)k * doutTotal + n] : 0.0f;
    Wt[i] = f2bf(v);
}

__global__ void k_fill(float* __restrict__ p, float v, long n) {
    long i = (long)blockIdx.x * blockDim.x + threadIdx.x;
    if (i < n) p[i] = v;
}

// C[M, ncols] = A[M, K](bf16) * Bt[npad, K](bf16 column fragments).
// One wave computes a 16 x (NT*16) strip: one A fragment is reused across NT
// independent accumulators -> NT independent v_wmma chains per k-step (ILP on
// the matrix pipe, A traffic / NT).
template<int NT>
__global__ void k_gemm_wmma(const unsigned short* __restrict__ A,
                            const unsigned short* __restrict__ Bt,
                            float* __restrict__ Cmat,
                            int M, int K, int ncols, int npad) {
    const int wavesPerBlock = blockDim.x >> 5;
    const int wave = blockIdx.x * wavesPerBlock + (threadIdx.x >> 5);
    const int lane = threadIdx.x & 31;
    const int groups = (npad >> 4) / NT;       // N-tile groups of NT
    const int tilesM = M >> 4;
    if (wave >= tilesM * groups) return;       // wave-uniform exit (EXEC all-1s for WMMA)
    const int tm = wave / groups;
    const int tg = wave % groups;

    const int row     = (tm << 4) + (lane & 15);          // A row for this lane
    const int colBase = tg * (NT << 4) + (lane & 15);     // first B column for this lane
    const int koff    = (lane >> 4) << 3;                 // 0 for lanes 0-15, 8 for lanes 16-31

    const uint4* __restrict__ Arow = (const uint4*)(A + (long)row * K);
    const uint4* Brow[NT];
    #pragma unroll
    for (int j = 0; j < NT; ++j)
        Brow[j] = (const uint4*)(Bt + (long)(colBase + (j << 4)) * K);

    v8f acc[NT];
    #pragma unroll
    for (int j = 0; j < NT; ++j) acc[j] = (v8f){};

    for (int kb = 0; kb < K; kb += 32) {
        const int i0 = (kb + koff) >> 3;                  // uint4 index (8 bf16 per uint4)
        FragBF a;
        a.u[0] = Arow[i0]; a.u[1] = Arow[i0 + 2];         // K = kb+koff..+7 , kb+16+koff..+7
        #pragma unroll
        for (int j = 0; j < NT; ++j) {
            FragBF b;
            b.u[0] = Brow[j][i0]; b.u[1] = Brow[j][i0 + 2];
            acc[j] = __builtin_amdgcn_wmma_f32_16x16x32_bf16(
                /*neg_a=*/false, a.v, /*neg_b=*/false, b.v,
                /*c_mod=*/(short)0, acc[j], /*reuse_a=*/false, /*reuse_b=*/false);
        }
    }

    const int mBase = (tm << 4) + ((lane >> 4) << 3);
    #pragma unroll
    for (int j = 0; j < NT; ++j) {
        const int col = colBase + (j << 4);
        if (col < ncols) {
            #pragma unroll
            for (int r = 0; r < 8; ++r)
                Cmat[(long)(mBase + r) * ncols + col] = acc[j][r];
        }
    }
}

// asn[n,h] = <h[n,h,:], a_s[h,:]> ; adn similarly
__global__ void k_attn_coef(const float* __restrict__ Hm,
                            const float* __restrict__ a_s, const float* __restrict__ a_d,
                            float* __restrict__ asn, float* __restrict__ adn, int dout) {
    int i = blockIdx.x * blockDim.x + threadIdx.x;
    if (i >= N_NODES * HEADS) return;
    int n = i / HEADS, h_ = i % HEADS;
    const int ld = HEADS * dout;
    const float* hp  = Hm + (long)n * ld + h_ * dout;
    const float* asp = a_s + h_ * dout;
    const float* adp = a_d + h_ * dout;
    float s1 = 0.0f, s2 = 0.0f;
    for (int c = 0; c < dout; ++c) { float v = hp[c]; s1 += v * asp[c]; s2 += v * adp[c]; }
    asn[i] = s1; adn[i] = s2;
}

__global__ void k_edge_max(const int* __restrict__ ei, const float* __restrict__ asn,
                           const float* __restrict__ adn, float* __restrict__ mbuf) {
    long i = (long)blockIdx.x * blockDim.x + threadIdx.x;
    if (i >= (long)ET_EDGES * HEADS) return;
    int e = (int)(i / HEADS), h_ = (int)(i % HEADS);
    int s, d; edge_nodes(ei, e, s, d);
    float v = lrelu(asn[s * HEADS + h_] + adn[d * HEADS + h_]);
    atomicMax(&mbuf[d * HEADS + h_], v);
}

__global__ void k_edge_sum(const int* __restrict__ ei, const float* __restrict__ asn,
                           const float* __restrict__ adn, const float* __restrict__ mbuf,
                           float* __restrict__ sbuf) {
    long i = (long)blockIdx.x * blockDim.x + threadIdx.x;
    if (i >= (long)ET_EDGES * HEADS) return;
    int e = (int)(i / HEADS), h_ = (int)(i % HEADS);
    int s, d; edge_nodes(ei, e, s, d);
    int di = d * HEADS + h_;
    float v = lrelu(asn[s * HEADS + h_] + adn[di]);
    atomicAdd(&sbuf[di], __expf(v - mbuf[di]));
}

// one block per (edge, head); blockDim = dout channels
__global__ void k_edge_aggr(const int* __restrict__ ei, const float* __restrict__ asn,
                            const float* __restrict__ adn, const float* __restrict__ mbuf,
                            const float* __restrict__ sbuf, const float* __restrict__ Hm,
                            float* __restrict__ outb, int dout) {
    const long bid = blockIdx.x;
    const int h_ = (int)(bid % HEADS);
    const int e  = (int)(bid / HEADS);
    int s, d; edge_nodes(ei, e, s, d);
    const int di = d * HEADS + h_;
    const float v = lrelu(asn[s * HEADS + h_] + adn[di]);
    const float alpha = __expf(v - mbuf[di]) / (sbuf[di] + 1e-16f);
    const int ld = HEADS * dout;
    const int c = threadIdx.x;
    atomicAdd(&outb[(long)d * ld + h_ * dout + c],
              alpha * Hm[(long)s * ld + h_ * dout + c]);
}

__global__ void k_finalize(const float* __restrict__ outb, const float* __restrict__ bias,
                           float* __restrict__ dst, int ld, long n) {
    long i = (long)blockIdx.x * blockDim.x + threadIdx.x;
    if (i >= n) return;
    float v = outb[i] + bias[i % ld];
    dst[i] = v > 0.0f ? v : 0.0f;
}

// ---------- launch ----------
static inline unsigned cdiv(long a, long b) { return (unsigned)((a + b - 1) / b); }

extern "C" void kernel_launch(void* const* d_in, const int* in_sizes, int n_in,
                              void* d_out, int out_size, void* d_ws, size_t ws_size,
                              hipStream_t stream) {
    (void)in_sizes; (void)n_in; (void)out_size; (void)ws_size;
    const float* x0 = (const float*)d_in[0];
    const int*   ei = (const int*)d_in[1];

    // workspace carving (256B aligned)
    char* ws = (char*)d_ws;
    size_t o = 0;
    auto carve = [&](size_t bytes) { void* p = ws + o; o = (o + bytes + 255) & ~(size_t)255; return p; };
    unsigned short* xbf  = (unsigned short*)carve((size_t)N_NODES * 256 * 2);       // bf16 layer input
    unsigned short* Wt   = (unsigned short*)carve((size_t)256 * 256 * 2);           // packed weights
    float* hbuf = (float*)carve((size_t)N_NODES * 256 * 4);                         // projected h
    float* asn  = (float*)carve((size_t)N_NODES * HEADS * 4);
    float* adn  = (float*)carve((size_t)N_NODES * HEADS * 4);
    float* mbuf = (float*)carve((size_t)N_NODES * HEADS * 4);
    float* sbuf = (float*)carve((size_t)N_NODES * HEADS * 4);
    float* outb = (float*)carve((size_t)N_NODES * 256 * 4);                         // aggregated output

    const int dins[5]  = {64, 256, 256, 256, 256};
    const int douts[5] = {64, 64, 64, 64, 10};

    const float* curx = x0;  // fp32 source for the per-layer bf16 cast

    for (int l = 0; l < 5; ++l) {
        const int din  = dins[l];
        const int dout = douts[l];
        const int dt   = HEADS * dout;                  // real output columns
        const int dpad = ((dt + 15) / 16) * 16;         // padded to 16
        const float* W  = (const float*)d_in[2 + 4 * l];
        const float* as = (const float*)d_in[3 + 4 * l];
        const float* ad = (const float*)d_in[4 + 4 * l];
        const float* b  = (const float*)d_in[5 + 4 * l];

        // 1. pack/transpose weights to bf16
        k_pack_w<<<cdiv((long)dpad * din, 256), 256, 0, stream>>>(W, Wt, din, dt, dpad);
        // 2. cast layer input to bf16
        k_cast_bf16<<<cdiv((long)N_NODES * din, 256), 256, 0, stream>>>(curx, xbf, (long)N_NODES * din);
        // 3. WMMA projection: hbuf[N, dt] = xbf * W
        if (dpad == 256) {   // layers 0-3: 16 N-tiles -> 4 groups of 4 per wave
            const int waves = (N_NODES / 16) * ((dpad / 16) / 4);
            k_gemm_wmma<4><<<cdiv(waves, 8), 256, 0, stream>>>(xbf, Wt, hbuf, N_NODES, din, dt, dpad);
        } else {             // layer 4: dpad=48 -> 3 N-tiles, one group of 3 per wave
            const int waves = (N_NODES / 16);
            k_gemm_wmma<3><<<cdiv(waves, 8), 256, 0, stream>>>(xbf, Wt, hbuf, N_NODES, din, dt, dpad);
        }
        // 4. attention coefficients
        k_attn_coef<<<cdiv((long)N_NODES * HEADS, 256), 256, 0, stream>>>(hbuf, as, ad, asn, adn, dout);
        // 5. init segment buffers
        k_fill<<<cdiv((long)N_NODES * HEADS, 256), 256, 0, stream>>>(mbuf, -1e30f, (long)N_NODES * HEADS);
        k_fill<<<cdiv((long)N_NODES * HEADS, 256), 256, 0, stream>>>(sbuf, 0.0f, (long)N_NODES * HEADS);
        k_fill<<<cdiv((long)N_NODES * dt, 256), 256, 0, stream>>>(outb, 0.0f, (long)N_NODES * dt);
        // 6-8. edge softmax + aggregation
        const long ne = (long)ET_EDGES * HEADS;
        k_edge_max<<<cdiv(ne, 256), 256, 0, stream>>>(ei, asn, adn, mbuf);
        k_edge_sum<<<cdiv(ne, 256), 256, 0, stream>>>(ei, asn, adn, mbuf, sbuf);
        k_edge_aggr<<<(unsigned)ne, dout, 0, stream>>>(ei, asn, adn, mbuf, sbuf, hbuf, outb, dout);
        // 9. bias + relu (in place for hidden layers, d_out for the last)
        float* dst = (l == 4) ? (float*)d_out : outb;
        k_finalize<<<cdiv((long)N_NODES * dt, 256), 256, 0, stream>>>(outb, b, dst, dt, (long)N_NODES * dt);

        curx = outb;
    }
}
